// BasicBlock_48722109006586
// MI455X (gfx1250) — matrix operand order repaired and checked by reading, not compiled, not run on hardware
//
#include <hip/hip_runtime.h>
#include <hip/hip_bf16.h>

typedef __attribute__((ext_vector_type(16))) _Float16 v16h;
typedef __attribute__((ext_vector_type(8)))  float    v8f;

#define B_N    32
#define CIN    256
#define HIN    64
#define WIN    64
#define COUT   512
#define OHW    32
#define KTOT   4096          // CIN * 4 * 4
#define KSTEP  32
#define NITER  (KTOT / KSTEP)
#define BM     64
#define BN     128
#define KTOP   410           // round(512 * 0.8)
#define BN_EPS 1e-5f
#define NEG_SLOPE 0.2f
#define THRESH_F 0.05f

// workspace byte offsets
#define WS_WMAX  0
#define WS_SUB   1024        // 32*256 f32   = 32 KiB
#define WS_SAL   65536       // 32*512 f32   = 64 KiB
#define WS_MASK  131072      // 32*512 f32   = 64 KiB
#define WS_MU    196608      // 512 f32
#define WS_ISTD  198656      // 512 f32
#define WS_TW    262144      // 4096*512 f16 = 4 MiB

// ---------------------------------------------------------------- init
__global__ void k_init(unsigned* wmax_bits) { *wmax_bits = 0u; }

// ------------------------------------------- sub[b,ci] = mean |x| over H*W
__global__ __launch_bounds__(256)
void k_absmean(const float* __restrict__ x, float* __restrict__ sub) {
  const int bc = blockIdx.x;                       // b*CIN + ci
  const float* p = x + (size_t)bc * (HIN * WIN);
  float s = 0.f;
  for (int i = threadIdx.x; i < HIN * WIN; i += 256) s += fabsf(p[i]);
  __shared__ float red[256];
  red[threadIdx.x] = s; __syncthreads();
  for (int off = 128; off; off >>= 1) {
    if (threadIdx.x < off) red[threadIdx.x] += red[threadIdx.x + off];
    __syncthreads();
  }
  if (threadIdx.x == 0) sub[bc] = red[0] * (1.f / (HIN * WIN));
}

// ------------------------------------------------------- max |W| reduction
__global__ __launch_bounds__(256)
void k_wmax(const float* __restrict__ w, unsigned* __restrict__ wmax_bits) {
  int base = blockIdx.x * 1024 + threadIdx.x;
  float m = 0.f;
  #pragma unroll
  for (int i = 0; i < 4; ++i) m = fmaxf(m, fabsf(w[base + 256 * i]));
  __shared__ float red[256];
  red[threadIdx.x] = m; __syncthreads();
  for (int off = 128; off; off >>= 1) {
    if (threadIdx.x < off) red[threadIdx.x] = fmaxf(red[threadIdx.x], red[threadIdx.x + off]);
    __syncthreads();
  }
  if (threadIdx.x == 0) atomicMax(wmax_bits, __float_as_uint(red[0]));
}

// ------------------- ternarize -> f16, repacked [k = ci*16+kh*4+kw][cout]
__global__ __launch_bounds__(256)
void k_ternarize(const float* __restrict__ w, const float* __restrict__ pos,
                 const float* __restrict__ neg, const unsigned* __restrict__ wmax_bits,
                 _Float16* __restrict__ tw) {
  const float t = THRESH_F * __uint_as_float(*wmax_bits);
  const float pv = pos[0], nv = neg[0];
  int idx = blockIdx.x * 1024 + threadIdx.x;
  #pragma unroll
  for (int i = 0; i < 4; ++i) {
    int id = idx + 256 * i;                        // co*4096 + ci*16 + r
    float wv = w[id];
    int co = id >> 12;
    int k  = id & 4095;                            // ci*16 + kh*4 + kw
    float q = (wv > t) ? pv : ((wv < -t) ? nv : 0.f);
    tw[(size_t)k * COUT + co] = (_Float16)q;
  }
}

// ----------------------------------- sal[b,co] = |sub[b,:]·sal_w[co,:]+b|
__global__ __launch_bounds__(256)
void k_saliency(const float* __restrict__ sub, const float* __restrict__ sal_w,
                const float* __restrict__ sal_b, float* __restrict__ sal) {
  int gid = blockIdx.x * 256 + threadIdx.x;        // < 32*512
  int b = gid >> 9, co = gid & 511;
  const float* sr = sub + b * CIN;
  const float* wr = sal_w + co * CIN;
  float acc = 0.f;
  for (int ci = 0; ci < CIN; ++ci) acc += sr[ci] * wr[ci];
  sal[gid] = fabsf(acc + sal_b[co]);
}

// --------------- exact k-th-largest per sample (stable rank), build mask
__global__ __launch_bounds__(512)
void k_threshold_mask(const float* __restrict__ sal, float* __restrict__ mask) {
  __shared__ float s[COUT];
  __shared__ float thr;
  const int b = blockIdx.x, t = threadIdx.x;
  s[t] = sal[b * COUT + t];
  __syncthreads();
  const float v = s[t];
  int rank = 0;
  for (int j = 0; j < COUT; ++j) {
    float u = s[j];
    rank += (u > v) || (u == v && j < t);
  }
  if (rank == KTOP - 1) thr = v;                   // exactly one writer
  __syncthreads();
  mask[b * COUT + t] = (v > thr) ? v : 0.f;
}

// ---------------------------------------------------- WMMA im2col GEMM conv
// grid (512, 4), block 256 (8 waves). M tile 64, N tile 128, K step 32.
// Double-buffered LDS pipeline; all gather address math hoisted out of the
// K loop (k&15 is slot-invariant => only the channel offset advances).
__global__ __launch_bounds__(256)
void k_conv_wmma(const float* __restrict__ x, const _Float16* __restrict__ tw,
                 const float* __restrict__ mask, float* __restrict__ y) {
  __shared__ v16h Alds[2][4][32];                  // [buf][msub][lane]  2x4KB
  __shared__ v16h Blds[2][8][32];                  // [buf][nsub][lane]  2x8KB

  const int tid  = threadIdx.x;
  const int lane = tid & 31;
  const int wave = tid >> 5;
  const int msub = wave & 3;
  const int nsb  = (wave >> 2) * 4;                // 0 or 4

  const int m_base = blockIdx.x * BM;              // output-position base
  const int n_base = blockIdx.y * BN;              // cout base
  const int bb     = m_base >> 10;                 // batch (fixed per block)

  // ---- precompute A staging slots: x index at kk=0, or -1 if padded ----
  int aoff[8];
  {
    const int j = tid & 15;
    const int v = j >> 1, h = j & 1;
    #pragma unroll
    for (int it = 0; it < 8; ++it) {
      int p = (tid >> 4) + 16 * it;                // fragment (msub,lane) pair
      int s = p >> 5;
      int L = p & 31;
      int k = ((v & 4) ? 16 : 0) + ((L >= 16) ? 8 : 0) + 2 * (v & 3) + h;
      int ci0 = k >> 4;                            // 0 or 1
      int rr  = k & 15;                            // invariant over K loop
      int kh = rr >> 2, kw = rr & 3;
      int m_g = m_base + 16 * s + (L & 15);
      int rem = m_g & 1023;
      int oh = rem >> 5, ow = rem & 31;
      int ih = 2 * oh - 1 + kh;
      int iw = 2 * ow - 1 + kw;
      bool ok = (unsigned)ih < (unsigned)HIN && (unsigned)iw < (unsigned)WIN;
      aoff[it] = ok ? ((bb * CIN + ci0) * HIN + ih) * WIN + iw : -1;
    }
  }
  // ---- precompute B staging slots: tw index at kk=0 ----
  int boff[16];
  {
    const int j = tid & 15;
    const int v = j >> 1, h = j & 1;
    #pragma unroll
    for (int it = 0; it < 16; ++it) {
      int p = (tid >> 4) + 16 * it;
      int s = p >> 5;                              // nsub
      int L = p & 31;
      int k = ((v & 4) ? 16 : 0) + ((L >= 16) ? 8 : 0) + 2 * (v & 3) + h;
      boff[it] = k * COUT + n_base + 16 * s + (L & 15);
    }
  }

  auto stage = [&](int buf, int kk) {
    _Float16* Ah = (_Float16*)Alds[buf];
    _Float16* Bh = (_Float16*)Blds[buf];
    const float*    xk  = x  + (size_t)kk * (CIN / 16 == 16 ? 256 : 256); // kk/16 * H*W
    const _Float16* twk = tw + (size_t)kk * COUT;
    // prefetch the A stream two K-steps ahead (global_prefetch_b8)
    if (aoff[0] >= 0 && kk + 2 * KSTEP < KTOT)
      __builtin_prefetch(xk + 2 * KSTEP * 256 + aoff[0], 0, 1);
    #pragma unroll
    for (int it = 0; it < 8; ++it) {
      float xv = (aoff[it] >= 0) ? xk[aoff[it]] : 0.f;
      Ah[tid + 256 * it] = (_Float16)xv;
    }
    #pragma unroll
    for (int it = 0; it < 16; ++it)
      Bh[tid + 256 * it] = twk[boff[it]];
  };

  v8f acc[4] = {};
  stage(0, 0);
  #pragma unroll 2
  for (int i = 0; i < NITER; ++i) {
    __syncthreads();                               // stage of buf (i&1) done
    if (i + 1 < NITER) stage((i + 1) & 1, (i + 1) * KSTEP);
    const int cur = i & 1;
    v16h a = Alds[cur][msub][lane];
    #pragma unroll
    for (int t = 0; t < 4; ++t) {
      v16h b = Blds[cur][nsb + t][lane];
      acc[t] = __builtin_amdgcn_wmma_f32_16x16x32_f16(
          false, a, false, b, (short)0, acc[t], false, false);
    }
  }

  // ---- epilogue: FBS mask multiply + store y (f32) ----
  #pragma unroll
  for (int t = 0; t < 4; ++t) {
    int n_g = n_base + 16 * (nsb + t) + (lane & 15);
    float mk = mask[bb * COUT + n_g];
    #pragma unroll
    for (int r = 0; r < 8; ++r) {
      int m_local = 16 * msub + r + ((lane >= 16) ? 8 : 0);
      int m_g = m_base + m_local;
      int rem = m_g & 1023;
      int oh = rem >> 5, ow = rem & 31;
      y[(((size_t)bb * COUT + n_g) * OHW + oh) * OHW + ow] = acc[t][r] * mk;
    }
  }
}

// --------------------------- per-channel batch stats (mean, 1/sqrt(var+eps))
__global__ __launch_bounds__(256)
void k_chanstats(const float* __restrict__ y, float* __restrict__ mu,
                 float* __restrict__ istd) {
  const int c = blockIdx.x;
  float s = 0.f, s2 = 0.f;
  for (int i = threadIdx.x; i < B_N * OHW * OHW; i += 256) {
    int b = i >> 10, sp = i & 1023;
    float v = y[(((size_t)b * COUT + c) << 10) + sp];
    s += v; s2 += v * v;
  }
  __shared__ float r1[256], r2[256];
  r1[threadIdx.x] = s; r2[threadIdx.x] = s2; __syncthreads();
  for (int off = 128; off; off >>= 1) {
    if (threadIdx.x < off) {
      r1[threadIdx.x] += r1[threadIdx.x + off];
      r2[threadIdx.x] += r2[threadIdx.x + off];
    }
    __syncthreads();
  }
  if (threadIdx.x == 0) {
    const float inv_n = 1.f / (B_N * OHW * OHW);
    float m = r1[0] * inv_n;
    float var = r2[0] * inv_n - m * m;
    mu[c]   = m;
    istd[c] = rsqrtf(var + BN_EPS);
  }
}

// ------------------------------------------- in-place BN + LeakyReLU(0.2)
__global__ __launch_bounds__(256)
void k_bn_lrelu(float* __restrict__ y, const float* __restrict__ mu,
                const float* __restrict__ istd, const float* __restrict__ gamma,
                const float* __restrict__ beta) {
  size_t base = (size_t)blockIdx.x * 2048 + threadIdx.x;
  #pragma unroll
  for (int i = 0; i < 8; ++i) {
    size_t idx = base + (size_t)i * 256;
    int c = (int)((idx >> 10) & 511);
    float v = y[idx];
    v = (v - mu[c]) * istd[c] * gamma[c] + beta[c];
    y[idx] = (v > 0.f) ? v : NEG_SLOPE * v;
  }
}

extern "C" void kernel_launch(void* const* d_in, const int* in_sizes, int n_in,
                              void* d_out, int out_size, void* d_ws, size_t ws_size,
                              hipStream_t stream) {
  const float* x      = (const float*)d_in[0];
  const float* weight = (const float*)d_in[1];
  const float* pos    = (const float*)d_in[2];
  const float* neg    = (const float*)d_in[3];
  const float* sal_w  = (const float*)d_in[4];
  const float* sal_b  = (const float*)d_in[5];
  const float* gamma  = (const float*)d_in[6];
  const float* beta   = (const float*)d_in[7];
  float* y = (float*)d_out;

  char* ws = (char*)d_ws;
  unsigned* wmax  = (unsigned*)(ws + WS_WMAX);
  float*    sub   = (float*)(ws + WS_SUB);
  float*    sal   = (float*)(ws + WS_SAL);
  float*    mask  = (float*)(ws + WS_MASK);
  float*    mu    = (float*)(ws + WS_MU);
  float*    istd  = (float*)(ws + WS_ISTD);
  _Float16* tw    = (_Float16*)(ws + WS_TW);

  k_init<<<1, 1, 0, stream>>>(wmax);
  k_absmean<<<B_N * CIN, 256, 0, stream>>>(x, sub);
  k_wmax<<<(COUT * CIN * 16) / 1024, 256, 0, stream>>>(weight, wmax);
  k_ternarize<<<(COUT * CIN * 16) / 1024, 256, 0, stream>>>(weight, pos, neg, wmax, tw);
  k_saliency<<<(B_N * COUT) / 256, 256, 0, stream>>>(sub, sal_w, sal_b, sal);
  k_threshold_mask<<<B_N, COUT, 0, stream>>>(sal, mask);
  dim3 cgrid((B_N * OHW * OHW) / BM, COUT / BN, 1);
  k_conv_wmma<<<cgrid, 256, 0, stream>>>(x, tw, mask, y);
  k_chanstats<<<COUT, 256, 0, stream>>>(y, mu, istd);
  k_bn_lrelu<<<(B_N * COUT * OHW * OHW) / 2048, 256, 0, stream>>>(y, mu, istd, gamma, beta);
}
